// LFADSG_22797686407628
// MI455X (gfx1250) — compile-verified, matrix operand order checked by reading, not run on hardware
//
#include <hip/hip_runtime.h>
#include <hip/hip_bf16.h>
#include <cstdint>

#define T_STEPS 100
#define NEUR    256
#define NBATCH  16
#define ROWS    4096      /* NBATCH*NEUR */
#define EPSF    1e-40f

typedef __attribute__((ext_vector_type(16))) _Float16 v16h;
typedef __attribute__((ext_vector_type(8)))  _Float16 v8h;
typedef __attribute__((ext_vector_type(8)))  float    v8f;
typedef __attribute__((ext_vector_type(4)))  unsigned int v4u;
typedef __attribute__((ext_vector_type(4)))  int v4i;
typedef __attribute__((ext_vector_type(8)))  int v8i;

static __device__ __forceinline__ v8f wmma16(v16h a, v16h b, v8f c) {
  // D = A(16x32 f16) * B(32x16 f16) + C(16x16 f32)
  return __builtin_amdgcn_wmma_f32_16x16x32_f16(false, a, false, b, (short)0, c, false, false);
}

// Load a 16x32 f16 A-fragment from a row-major [16+,ld] tile (also used for
// B fragments when the weight matrix is stored N-major, i.e. [N][K]).
// ISA layout: lane<16 -> row=lane, K = {0..7,16..23}; lane>=16 -> row=lane-16,
// K = {8..15,24..31}.  All loads are 16B aligned (ld multiple of 8 halves).
static __device__ __forceinline__ v16h frag16(const _Float16* base, int ld, int lane) {
  const int r  = lane & 15;
  const int kb = (lane >> 4) << 3;
  const _Float16* p = base + r * ld + kb;
  v8h lo = *(const v8h*)(p);
  v8h hi = *(const v8h*)(p + 16);
  return __builtin_shufflevector(lo, hi, 0,1,2,3,4,5,6,7,8,9,10,11,12,13,14,15);
}

static __device__ __forceinline__ uint32_t hash3(uint32_t a, uint32_t b, uint32_t c) {
  a *= 0xcc9e2d51u; a ^= a >> 17;
  a += b * 0x1b873593u; a ^= a >> 13;
  a *= 0x9E3779B9u; a ^= c * 0x85ebca6bu;
  a ^= a >> 16;
  return a;
}
static __device__ __forceinline__ float u01(uint32_t h) {
  return (float)(h >> 8) * (1.0f / 16777216.0f) + 1e-7f;
}
// Fast transcendentals: avoid the IEEE v_div_scale/v_rcp refinement chains in
// the WMMA epilogues -- v_rcp_f32 + v_exp_f32 only.
static __device__ __forceinline__ float fast_rcp(float x) { return __builtin_amdgcn_rcpf(x); }
static __device__ __forceinline__ float sigmoidf(float v) { return fast_rcp(1.f + __expf(-v)); }
static __device__ __forceinline__ float tanh_fast(float v) {
  float e = __expf(-2.f * fabsf(v));
  float t = (1.f - e) * fast_rcp(1.f + e);
  return v < 0.f ? -t : t;
}

// ---------------------------------------------------------------------------
// Tensor Data Mover: 1-row 2D descriptor moving `bytes` (f16 data) from global
// to LDS.  D# bitfields per CDNA5 ISA.  elems < 65536.
// ---------------------------------------------------------------------------
static __device__ __forceinline__ void tdm_load_1d(void* lds_dst, const void* gsrc,
                                                   unsigned bytes) {
  unsigned elems = bytes >> 1;  // data_size = 2 bytes
  unsigned long long ga = (unsigned long long)(size_t)gsrc;
  unsigned lds = (unsigned)(size_t)lds_dst;  // addr[31:0] == LDS byte offset
  v4u g0;
  g0[0] = 1u;                                          // count=1 valid user D#
  g0[1] = lds;                                         // lds_addr
  g0[2] = (unsigned)(ga & 0xffffffffu);                // global_addr[31:0]
  g0[3] = (unsigned)((ga >> 32) & 0x01ffffffu) | (2u << 30);  // addr[56:32]+type=2
  v8i g1;
  g1[0] = (int)(1u << 16);                  // wg_mask=0, data_size=1 (2B)
  g1[1] = (int)((elems & 0xffffu) << 16);   // tensor_dim0[15:0] -> bits 63:48
  g1[2] = (int)(((elems >> 16) & 0xffffu)   // tensor_dim0[31:16] -> bits 79:64
                | (1u << 16));              // tensor_dim1 = 1 -> bits 95:80
  g1[3] = (int)(elems << 16);               // tile_dim0 -> bits 127:112
  g1[4] = 1;                                // tile_dim1 = 1 (tile_dim2 = 0)
  g1[5] = (int)elems;                       // tensor_dim0_stride[31:0]
  g1[6] = 0;                                // stride hi + dim1_stride lo
  g1[7] = 0;
  v4i gz = {0, 0, 0, 0};
#if __clang_major__ >= 23
  v8i gz8 = {0, 0, 0, 0, 0, 0, 0, 0};
  __builtin_amdgcn_tensor_load_to_lds(g0, g1, gz, gz, gz8, 0);
#else
  __builtin_amdgcn_tensor_load_to_lds(g0, g1, gz, gz, 0);
#endif
}

// ---------------------------------------------------------------------------
// Weight convert: dst[n*K+k] = (f16) src[(k+row_off)*N + n]   (N-major f16)
// ---------------------------------------------------------------------------
__global__ void wconv_kernel(const float* __restrict__ src, _Float16* __restrict__ dst,
                             int K, int N, int row_off) {
  int idx = blockIdx.x * 256 + threadIdx.x;
  if (idx >= K * N) return;
  int n = idx / K, k = idx % K;
  dst[idx] = (_Float16)src[(size_t)(k + row_off) * N + n];
}

// ---------------------------------------------------------------------------
// Encoder GRU (din=1, dh=64).  Block = 128 threads = 4 waves, 64 rows resident.
// h lives in LDS for all 100 steps; weights staged once via the TDM.
// Per wave per step: ru GEMM (8 ntiles x 2 ktiles) + c GEMM (4 x 2) = 24 WMMAs.
// ---------------------------------------------------------------------------
__global__ __launch_bounds__(128) void enc_gru_kernel(
    const float* __restrict__ inputs,   // [B,T,N] f32
    const _Float16* __restrict__ wruT,  // [128][64] f16 (rows 1..64 of Wru, N-major)
    const float* __restrict__ wru0,     // [128] = Wru row 0 (x weight)
    const float* __restrict__ bru,      // [128]
    const _Float16* __restrict__ wcT,   // [64][64]
    const float* __restrict__ wc0,      // [64] = Wc row 0
    const float* __restrict__ bc,       // [64]
    _Float16* __restrict__ hist,        // [T][4096][64] f16 or null
    _Float16* __restrict__ g0dst,       // [4096][128] (+col offset baked) or null
    int backward) {
  __shared__ __align__(16) _Float16 wru_s[128 * 64];
  __shared__ __align__(16) _Float16 wc_s[64 * 64];
  __shared__ __align__(16) _Float16 h_s[64][64];
  __shared__ __align__(16) _Float16 rh_s[64][64];
  __shared__ float u_s[64][64];
  __shared__ float x_s[64];

  const int tid = threadIdx.x;
  const int lane = tid & 31;
  const int wave = tid >> 5;
  const int row0 = blockIdx.x * 64;
  const int mb = wave * 16;

  // Async weight staging through the Tensor Data Mover (TENSORcnt tracked).
  if (tid == 0) {
    tdm_load_1d(wru_s, wruT, 128 * 64 * 2);
    tdm_load_1d(wc_s, wcT, 64 * 64 * 2);
  }
  for (int i = tid; i < 64 * 64; i += 128) (&h_s[0][0])[i] = (_Float16)0.f;
  if (tid == 0) __builtin_amdgcn_s_wait_tensorcnt(0);
  __syncthreads();

  for (int s = 0; s < T_STEPS; ++s) {
    const int t = backward ? (T_STEPS - 1 - s) : s;
    if (lane < 16) {
      int r = row0 + mb + lane;
      int b = r >> 8, n = r & 255;
      x_s[mb + lane] = inputs[((size_t)b * T_STEPS + t) * NEUR + n];
      if (s + 1 < T_STEPS) {  // prefetch next step's x (global_prefetch_b8)
        int tn = backward ? (t - 1) : (t + 1);
        __builtin_prefetch(&inputs[((size_t)b * T_STEPS + tn) * NEUR + n], 0, 1);
      }
    }
    __syncthreads();

    // GEMM 1: ru = sigmoid([x|h] @ Wru + bru)   (K split: x rank-1 + h@WruT)
    for (int nt = 0; nt < 8; ++nt) {
      v8f acc = {};
#pragma unroll
      for (int kt = 0; kt < 2; ++kt) {
        v16h a = frag16(&h_s[mb][0] + kt * 32, 64, lane);
        v16h b = frag16(wru_s + (nt * 16) * 64 + kt * 32, 64, lane);
        acc = wmma16(a, b, acc);
      }
      int n = nt * 16 + (lane & 15);
      int mh = mb + ((lane >> 4) << 3);
      float w0 = wru0[n], bi = bru[n];
#pragma unroll
      for (int i = 0; i < 8; ++i) {
        int m = mh + i;
        float sg = sigmoidf(acc[i] + x_s[m] * w0 + bi);
        if (n < 64) rh_s[m][n] = (_Float16)(sg * (float)h_s[m][n]);
        else        u_s[m][n - 64] = sg;
      }
    }
    __syncthreads();

    // GEMM 2: c = tanh([x|r*h] @ Wc + bc);  h = u*h + (1-u)*c
    for (int nt = 0; nt < 4; ++nt) {
      v8f acc = {};
#pragma unroll
      for (int kt = 0; kt < 2; ++kt) {
        v16h a = frag16(&rh_s[mb][0] + kt * 32, 64, lane);
        v16h b = frag16(wc_s + (nt * 16) * 64 + kt * 32, 64, lane);
        acc = wmma16(a, b, acc);
      }
      int n = nt * 16 + (lane & 15);
      int mh = mb + ((lane >> 4) << 3);
      float w0 = wc0[n], bi = bc[n];
#pragma unroll
      for (int i = 0; i < 8; ++i) {
        int m = mh + i;
        float c = tanh_fast(acc[i] + x_s[m] * w0 + bi);
        float u = u_s[m][n];
        float hn = u * (float)h_s[m][n] + (1.f - u) * c;
        h_s[m][n] = (_Float16)hn;
        if (hist) hist[((size_t)t * ROWS + row0 + m) * 64 + n] = (_Float16)hn;
      }
    }
    __syncthreads();
  }

  if (g0dst) {
    for (int idx = tid; idx < 64 * 64; idx += 128) {
      int m = idx >> 6, d = idx & 63;
      g0dst[(size_t)(row0 + m) * 128 + d] = h_s[m][d];
    }
  }
}

// ---------------------------------------------------------------------------
// g0_prob = sigmoid(in_g0 @ in_g0^T) per batch (256x128 x 128x256).
// Grid: 16 batches * 16 mtiles; wave w covers ntiles 4w..4w+3.
// ---------------------------------------------------------------------------
__global__ __launch_bounds__(128) void g0prob_kernel(
    const _Float16* __restrict__ ing0,  // [4096][128]
    float* __restrict__ prob) {         // [16][256][256]
  int bid = blockIdx.x;
  int b = bid >> 4, mt = bid & 15;
  int wave = threadIdx.x >> 5, lane = threadIdx.x & 31;
  const _Float16* Ab = ing0 + (size_t)b * NEUR * 128;
  for (int q = 0; q < 4; ++q) {
    int nt = wave * 4 + q;
    v8f acc = {};
#pragma unroll
    for (int kt = 0; kt < 4; ++kt) {
      v16h a = frag16(Ab + (mt * 16) * 128 + kt * 32, 128, lane);
      v16h bf = frag16(Ab + (nt * 16) * 128 + kt * 32, 128, lane);  // B = A^T, N-major == A
      acc = wmma16(a, bf, acc);
    }
    int n = nt * 16 + (lane & 15);
    int mh = mt * 16 + ((lane >> 4) << 3);
#pragma unroll
    for (int i = 0; i < 8; ++i)
      prob[((size_t)b * NEUR + mh + i) * NEUR + n] = sigmoidf(acc[i]);
  }
}

// ---------------------------------------------------------------------------
// Row sums of a [4096][256] f32 matrix (one block per row).
// ---------------------------------------------------------------------------
__global__ void rowsum_kernel(const float* __restrict__ g, float* __restrict__ rs) {
  __shared__ float red[256];
  int row = blockIdx.x, j = threadIdx.x;
  red[j] = g[(size_t)row * NEUR + j];
  __syncthreads();
  for (int s = 128; s > 0; s >>= 1) {
    if (j < s) red[j] += red[j + s];
    __syncthreads();
  }
  if (!j) rs[row] = red[0];
}

// out16[i] = (f16)( g[i] * rsqrt(rs[row]) * rsqrt(rs[col-of-same-batch]) )
__global__ void normalize_f16_kernel(const float* __restrict__ g, const float* __restrict__ rs,
                                     _Float16* __restrict__ out) {
  size_t i = (size_t)blockIdx.x * 256 + threadIdx.x;  // 16*256*256 elems
  int brow = (int)(i >> 8);
  int j = (int)(i & 255);
  int b = brow >> 8;
  float v = g[i] * rsqrtf(rs[brow]) * rsqrtf(rs[(b << 8) + j]);
  out[i] = (_Float16)v;
}

// ---------------------------------------------------------------------------
// Batched graph GEMM: C[b] = G[b](256x256,f16) @ B[b](256x64) [+bias][tanh].
// B supplied transposed (N-major [16][64][256]).  Grid 16*16 blocks, wave=ntile.
// ---------------------------------------------------------------------------
__global__ __launch_bounds__(128) void graph_gemm_kernel(
    const _Float16* __restrict__ G,   // [16][256][256]
    const _Float16* __restrict__ BT,  // [16][64][256]
    const float* __restrict__ bias,   // [64] or null
    float* __restrict__ outF32,       // [4096][64] or null
    _Float16* __restrict__ outRM,     // [4096][64] row-major f16 or null
    int apply_tanh) {
  int bid = blockIdx.x;
  int b = bid >> 4, mt = bid & 15;
  int wave = threadIdx.x >> 5, lane = threadIdx.x & 31;
  const _Float16* Gb = G + (size_t)b * NEUR * NEUR;
  const _Float16* Bb = BT + (size_t)b * 64 * NEUR;
  int nt = wave;
  v8f acc = {};
#pragma unroll
  for (int kt = 0; kt < 8; ++kt) {
    v16h a = frag16(Gb + (mt * 16) * NEUR + kt * 32, NEUR, lane);
    v16h bf = frag16(Bb + (nt * 16) * NEUR + kt * 32, NEUR, lane);
    acc = wmma16(a, bf, acc);
  }
  int n = nt * 16 + (lane & 15);
  int mh = mt * 16 + ((lane >> 4) << 3);
  float bi = bias ? bias[n] : 0.f;
#pragma unroll
  for (int i = 0; i < 8; ++i) {
    int row = b * NEUR + mh + i;
    float v = acc[i] + bi;
    if (apply_tanh) v = tanh_fast(v);
    if (outF32) outF32[(size_t)row * 64 + n] = v;
    if (outRM) outRM[(size_t)row * 64 + n] = (_Float16)v;
  }
}

// ---------------------------------------------------------------------------
// Fused decoder GRU step + reparameterized latent.
// xh = [e2f[t-1] | e2b[t+1] | h_mp]  (192 wide), 64 rows per block.
// ---------------------------------------------------------------------------
__global__ __launch_bounds__(128) void grnn_step_kernel(
    const _Float16* __restrict__ e2f,  // [T][4096][64]
    const _Float16* __restrict__ e2b,  // [T][4096][64]
    const float* __restrict__ hmp,     // [4096][64]
    const _Float16* __restrict__ wruT, // [128][192]
    const float* __restrict__ bru,
    const _Float16* __restrict__ wcT,  // [64][192]
    const float* __restrict__ bc,
    const _Float16* __restrict__ WmT, const float* __restrict__ bm,
    const _Float16* __restrict__ WvT, const float* __restrict__ bv,
    _Float16* __restrict__ hT16,       // [16][64][256] new hidden (transposed)
    _Float16* __restrict__ x16,        // [4096][64] latent, row-major f16
    int t) {
  __shared__ __align__(16) _Float16 xh_s[64][192];
  __shared__ float u_s[64][64];
  __shared__ _Float16 r_s[64][64];
  __shared__ __align__(16) _Float16 hn_s[64][64];

  int tid = threadIdx.x, lane = tid & 31, wave = tid >> 5;
  int row0 = blockIdx.x * 64;
  int mb = wave * 16;

  for (int idx = tid; idx < 64 * 192; idx += 128) {
    int m = idx / 192, k = idx % 192;
    int row = row0 + m;
    _Float16 v;
    if (k < 64)
      v = (t >= 1) ? e2f[((size_t)(t - 1) * ROWS + row) * 64 + k] : (_Float16)0.f;
    else if (k < 128)
      v = (t + 1 < T_STEPS) ? e2b[((size_t)(t + 1) * ROWS + row) * 64 + (k - 64)] : (_Float16)0.f;
    else
      v = (_Float16)hmp[(size_t)row * 64 + (k - 128)];
    xh_s[m][k] = v;
  }
  __syncthreads();

  // GEMM 1: ru = sigmoid(xh @ Wru + bru)   8 ntiles x 6 ktiles
  for (int nt = 0; nt < 8; ++nt) {
    v8f acc = {};
#pragma unroll
    for (int kt = 0; kt < 6; ++kt) {
      v16h a = frag16(&xh_s[mb][0] + kt * 32, 192, lane);
      v16h bf = frag16(wruT + (nt * 16) * 192 + kt * 32, 192, lane);
      acc = wmma16(a, bf, acc);
    }
    int n = nt * 16 + (lane & 15);
    int mh = mb + ((lane >> 4) << 3);
    float bi = bru[n];
#pragma unroll
    for (int i = 0; i < 8; ++i) {
      int m = mh + i;
      float sg = sigmoidf(acc[i] + bi);
      if (n < 64) r_s[m][n] = (_Float16)sg;
      else        u_s[m][n - 64] = sg;
    }
  }
  __syncthreads();

  // Apply reset gate in-place: xh[:,128:] = r * h_mp (deferred so GEMM1 saw clean A)
  for (int idx = tid; idx < 64 * 64; idx += 128) {
    int m = idx >> 6, n = idx & 63;
    xh_s[m][128 + n] = (_Float16)((float)xh_s[m][128 + n] * (float)r_s[m][n]);
  }
  __syncthreads();

  // GEMM 2: c = tanh(xh' @ Wc + bc); h_new = u*h_mp + (1-u)*c
  for (int nt = 0; nt < 4; ++nt) {
    v8f acc = {};
#pragma unroll
    for (int kt = 0; kt < 6; ++kt) {
      v16h a = frag16(&xh_s[mb][0] + kt * 32, 192, lane);
      v16h bf = frag16(wcT + (nt * 16) * 192 + kt * 32, 192, lane);
      acc = wmma16(a, bf, acc);
    }
    int n = nt * 16 + (lane & 15);
    int mh = mb + ((lane >> 4) << 3);
    float bi = bc[n];
#pragma unroll
    for (int i = 0; i < 8; ++i) {
      int m = mh + i;
      float c = tanh_fast(acc[i] + bi);
      float u = u_s[m][n];
      float h0 = hmp[(size_t)(row0 + m) * 64 + n];
      float hn = u * h0 + (1.f - u) * c;
      hn_s[m][n] = (_Float16)hn;
      int row = row0 + m, b = row >> 8, j = row & 255;
      hT16[((size_t)b * 64 + n) * NEUR + j] = (_Float16)hn;
    }
  }
  __syncthreads();

  // GEMM 3: mean/logvar + reparameterized sample  (4 ntiles x 2 ktiles x 2 mats)
  for (int nt = 0; nt < 4; ++nt) {
    v8f am = {}, av = {};
#pragma unroll
    for (int kt = 0; kt < 2; ++kt) {
      v16h a = frag16(&hn_s[mb][0] + kt * 32, 64, lane);
      v16h bmf = frag16(WmT + (nt * 16) * 64 + kt * 32, 64, lane);
      v16h bvf = frag16(WvT + (nt * 16) * 64 + kt * 32, 64, lane);
      am = wmma16(a, bmf, am);
      av = wmma16(a, bvf, av);
    }
    int n = nt * 16 + (lane & 15);
    int mh = mb + ((lane >> 4) << 3);
    float bmv = bm[n], bvv = bv[n];
#pragma unroll
    for (int i = 0; i < 8; ++i) {
      int row = row0 + mh + i;
      float mean = am[i] + bmv, lv = av[i] + bvv;
      uint32_t h1 = hash3(0xA511E9B3u + (uint32_t)t * 2654435761u, (uint32_t)row, (uint32_t)n);
      uint32_t h2 = hash3(0x5BD1E995u + (uint32_t)t * 2246822519u, (uint32_t)row, (uint32_t)n);
      float eps = __fsqrt_rn(-2.f * __logf(u01(h1))) * __cosf(6.2831853f * u01(h2));
      float x = mean + __expf(0.5f * lv) * eps;
      x16[(size_t)row * 64 + n] = (_Float16)x;
    }
  }
}

// ---------------------------------------------------------------------------
// Per-step symmetric Gumbel graph sample + row sums (block per (b,row)).
// ---------------------------------------------------------------------------
__global__ void graph_sample_kernel(const float* __restrict__ prob, float* __restrict__ g,
                                    float* __restrict__ rs, int t) {
  __shared__ float red[256];
  int brow = blockIdx.x, j = threadIdx.x;
  int i = brow & 255;
  float p = prob[(size_t)brow * NEUR + j];
  int lo = i < j ? i : j, hi = i < j ? j : i;
  uint32_t hA = hash3(0x01234567u + (uint32_t)t * 2654435761u, (uint32_t)lo, (uint32_t)hi);
  uint32_t hB = hash3(0x89ABCDEFu + (uint32_t)t * 2246822519u, (uint32_t)lo, (uint32_t)hi);
  float n1 = -__logf(-__logf(u01(hA) + EPSF) + EPSF);
  float n2 = -__logf(-__logf(u01(hB) + EPSF) + EPSF);
  float v = (__logf(p + EPSF) + n1 - (__logf(1.f - p + EPSF) + n2)) * 2.0f;  // /TAU, TAU=0.5
  float s = sigmoidf(v);
  g[(size_t)brow * NEUR + j] = s;
  red[j] = s;
  __syncthreads();
  for (int k = 128; k > 0; k >>= 1) {
    if (j < k) red[j] += red[j + k];
    __syncthreads();
  }
  if (!j) rs[brow] = red[0];
}

// ---------------------------------------------------------------------------
// Dense 64x64: out^T[b][n][row] = (f16)(A16 @ W)   (transposed for graph GEMM B)
// ---------------------------------------------------------------------------
__global__ __launch_bounds__(128) void dense64T_kernel(
    const _Float16* __restrict__ A16,  // [4096][64]
    const _Float16* __restrict__ WT,   // [64][64] N-major
    _Float16* __restrict__ outT) {     // [16][64][256]
  int wave = threadIdx.x >> 5, lane = threadIdx.x & 31;
  int row0 = blockIdx.x * 64;
  int mb = wave * 16;
  for (int nt = 0; nt < 4; ++nt) {
    v8f acc = {};
#pragma unroll
    for (int kt = 0; kt < 2; ++kt) {
      v16h a = frag16(A16 + (size_t)(row0 + mb) * 64 + kt * 32, 64, lane);
      v16h b = frag16(WT + (nt * 16) * 64 + kt * 32, 64, lane);
      acc = wmma16(a, b, acc);
    }
    int n = nt * 16 + (lane & 15);
    int mh = mb + ((lane >> 4) << 3);
#pragma unroll
    for (int i = 0; i < 8; ++i) {
      int row = row0 + mh + i, b_ = row >> 8, j = row & 255;
      outT[((size_t)b_ * 64 + n) * NEUR + j] = (_Float16)acc[i];
    }
  }
}

// Output head: rate = (tanh(x@Wout+bout)+1)/2 + EPS  (exp(log(x+eps)) == x+eps)
__global__ void output_kernel(const _Float16* __restrict__ x16, const float* __restrict__ wout,
                              const float* __restrict__ bout, float* __restrict__ out, int t) {
  int row = blockIdx.x * 256 + threadIdx.x;  // 4096 rows
  float acc = bout[0];
  const _Float16* xr = x16 + (size_t)row * 64;
#pragma unroll
  for (int d = 0; d < 64; ++d) acc += (float)xr[d] * wout[d];
  float v = (tanh_fast(acc) + 1.f) * 0.5f + EPSF;
  int b = row >> 8, n = row & 255;
  out[((size_t)b * T_STEPS + t) * NEUR + n] = v;
}

// ===========================================================================
extern "C" void kernel_launch(void* const* d_in, const int* in_sizes, int n_in,
                              void* d_out, int out_size, void* d_ws, size_t ws_size,
                              hipStream_t stream) {
  (void)in_sizes; (void)n_in; (void)out_size; (void)ws_size;

  const float* inputs = (const float*)d_in[0];
  // enc params: dict order Wru,bru,Wc,bc ; GRUs at base 1+4*g (enc1f,enc1b,enc2f,enc2b)
  const float* encWru[4]; const float* encBru[4]; const float* encWc[4]; const float* encBc[4];
  for (int g = 0; g < 4; ++g) {
    encWru[g] = (const float*)d_in[1 + 4 * g + 0];
    encBru[g] = (const float*)d_in[1 + 4 * g + 1];
    encWc[g]  = (const float*)d_in[1 + 4 * g + 2];
    encBc[g]  = (const float*)d_in[1 + 4 * g + 3];
  }
  const float* gWru = (const float*)d_in[17];
  const float* gBru = (const float*)d_in[18];
  const float* gWc  = (const float*)d_in[19];
  const float* gBc  = (const float*)d_in[20];
  const float* Wm = (const float*)d_in[21]; const float* bm = (const float*)d_in[22];
  const float* Wv = (const float*)d_in[23]; const float* bv = (const float*)d_in[24];
  const float* g0W = (const float*)d_in[25]; const float* g0b = (const float*)d_in[26];
  const float* g1W = (const float*)d_in[27]; const float* g1b = (const float*)d_in[28];
  const float* Wout = (const float*)d_in[29]; const float* bout = (const float*)d_in[30];
  float* out = (float*)d_out;

  // -------- workspace bump allocator (256B aligned) --------
  char* ws = (char*)d_ws;
  size_t off = 0;
  auto alloc = [&](size_t bytes) -> void* {
    void* p = ws + off;
    off = (off + bytes + 255) & ~(size_t)255;
    return p;
  };
  _Float16* encWruT[4]; _Float16* encWcT[4];
  for (int g = 0; g < 4; ++g) {
    encWruT[g] = (_Float16*)alloc(128 * 64 * 2);
    encWcT[g]  = (_Float16*)alloc(64 * 64 * 2);
  }
  _Float16* gWruT = (_Float16*)alloc(128 * 192 * 2);
  _Float16* gWcT  = (_Float16*)alloc(64 * 192 * 2);
  _Float16* WmT = (_Float16*)alloc(64 * 64 * 2);
  _Float16* WvT = (_Float16*)alloc(64 * 64 * 2);
  _Float16* G0T = (_Float16*)alloc(64 * 64 * 2);
  _Float16* G1T = (_Float16*)alloc(64 * 64 * 2);
  _Float16* E2F  = (_Float16*)alloc((size_t)T_STEPS * ROWS * 64 * 2);
  _Float16* E2B  = (_Float16*)alloc((size_t)T_STEPS * ROWS * 64 * 2);
  _Float16* ING0 = (_Float16*)alloc((size_t)ROWS * 128 * 2);
  float* PROB = (float*)alloc((size_t)NBATCH * NEUR * NEUR * 4);
  float* RS0  = (float*)alloc(ROWS * 4);
  _Float16* GPN16 = (_Float16*)alloc((size_t)NBATCH * NEUR * NEUR * 2);
  float* GRAPH = (float*)alloc((size_t)NBATCH * NEUR * NEUR * 4);
  float* RS1  = (float*)alloc(ROWS * 4);
  _Float16* GRAPH16 = (_Float16*)alloc((size_t)NBATCH * NEUR * NEUR * 2);
  _Float16* HT16 = (_Float16*)alloc((size_t)NBATCH * 64 * NEUR * 2);
  float* HMP = (float*)alloc((size_t)ROWS * 64 * 4);
  _Float16* X16a = (_Float16*)alloc((size_t)ROWS * 64 * 2);
  _Float16* X16b = (_Float16*)alloc((size_t)ROWS * 64 * 2);
  _Float16* XWT16 = (_Float16*)alloc((size_t)NBATCH * 64 * NEUR * 2);

  // -------- weight conversion (f32 -> f16, N-major) --------
  auto wconv = [&](const float* src, _Float16* dst, int K, int N, int row_off) {
    int total = K * N;
    wconv_kernel<<<(total + 255) / 256, 256, 0, stream>>>(src, dst, K, N, row_off);
  };
  for (int g = 0; g < 4; ++g) {
    wconv(encWru[g], encWruT[g], 64, 128, 1);  // skip x-row (row 0)
    wconv(encWc[g],  encWcT[g],  64, 64, 1);
  }
  wconv(gWru, gWruT, 192, 128, 0);
  wconv(gWc,  gWcT,  192, 64, 0);
  wconv(Wm, WmT, 64, 64, 0);
  wconv(Wv, WvT, 64, 64, 0);
  wconv(g0W, G0T, 64, 64, 0);
  wconv(g1W, G1T, 64, 64, 0);

  // h0 for decoder GRU = 0
  hipMemsetAsync(HT16, 0, (size_t)NBATCH * 64 * NEUR * 2, stream);

  // -------- encoders (persistent-state GRUs) --------
  // enc1f/enc1b -> in_g0 (final hidden states), enc2f/enc2b -> full histories
  enc_gru_kernel<<<64, 128, 0, stream>>>(inputs, encWruT[0], encWru[0], encBru[0],
                                         encWcT[0], encWc[0], encBc[0],
                                         nullptr, ING0, 0);
  enc_gru_kernel<<<64, 128, 0, stream>>>(inputs, encWruT[1], encWru[1], encBru[1],
                                         encWcT[1], encWc[1], encBc[1],
                                         nullptr, ING0 + 64, 1);
  enc_gru_kernel<<<64, 128, 0, stream>>>(inputs, encWruT[2], encWru[2], encBru[2],
                                         encWcT[2], encWc[2], encBc[2],
                                         E2F, nullptr, 0);
  enc_gru_kernel<<<64, 128, 0, stream>>>(inputs, encWruT[3], encWru[3], encBru[3],
                                         encWcT[3], encWc[3], encBc[3],
                                         E2B, nullptr, 1);

  // -------- static graph probability + normalized (loop-invariant) --------
  g0prob_kernel<<<NBATCH * 16, 128, 0, stream>>>(ING0, PROB);
  rowsum_kernel<<<ROWS, 256, 0, stream>>>(PROB, RS0);
  normalize_f16_kernel<<<(NBATCH * NEUR * NEUR) / 256, 256, 0, stream>>>(PROB, RS0, GPN16);

  // -------- decoder scan over time --------
  for (int t = 0; t < T_STEPS; ++t) {
    // h_mp = gp_norm @ h
    graph_gemm_kernel<<<NBATCH * 16, 128, 0, stream>>>(GPN16, HT16, nullptr, HMP, nullptr, 0);
    // GRU step + mean/logvar + reparam latent
    grnn_step_kernel<<<64, 128, 0, stream>>>(E2F, E2B, HMP, gWruT, gBru, gWcT, gBc,
                                             WmT, bm, WvT, bv, HT16, X16a, t);
    // per-step sampled + normalized graph
    graph_sample_kernel<<<ROWS, 256, 0, stream>>>(PROB, GRAPH, RS1, t);
    normalize_f16_kernel<<<(NBATCH * NEUR * NEUR) / 256, 256, 0, stream>>>(GRAPH, RS1, GRAPH16);
    // GCN layer 0: x = tanh(graph @ (x @ W0) + b0)
    dense64T_kernel<<<64, 128, 0, stream>>>(X16a, G0T, XWT16);
    graph_gemm_kernel<<<NBATCH * 16, 128, 0, stream>>>(GRAPH16, XWT16, g0b, nullptr, X16b, 1);
    // GCN layer 1
    dense64T_kernel<<<64, 128, 0, stream>>>(X16b, G1T, XWT16);
    graph_gemm_kernel<<<NBATCH * 16, 128, 0, stream>>>(GRAPH16, XWT16, g1b, nullptr, X16a, 1);
    // output head -> rates[b][t][n]
    output_kernel<<<ROWS / 256, 256, 0, stream>>>(X16a, Wout, bout, out, t);
  }
}
